// BatchHardTripletBlendLoss_77678778515586
// MI455X (gfx1250) — matrix operand order repaired and checked by reading, not compiled, not run on hardware
//
#include <hip/hip_runtime.h>

typedef unsigned int u32;
typedef unsigned long long u64;
typedef unsigned short ushort_t;
typedef __attribute__((ext_vector_type(4)))  u32    v4u;
typedef __attribute__((ext_vector_type(16))) __bf16 v16bf;
typedef __attribute__((ext_vector_type(8)))  float  v8f;

// 128-bit payload type + address-space-qualified pointers matching the async builtin.
typedef int v4i_vs __attribute__((vector_size(16)));
typedef __attribute__((address_space(1))) v4i_vs* as1_v4i;
typedef __attribute__((address_space(3))) v4i_vs* as3_v4i;

#define KC  32      // WMMA K per chunk (bf16 16x16x32)
#define LDT 40      // LDS row pitch in ushorts (80B: 16B aligned, conflict-free ds_read_b128)

// ---- CDNA5 async global->LDS copy (ASYNCcnt path) ----
__device__ __forceinline__ void async_ld_b128(const ushort_t* g, ushort_t* l) {
#if __has_builtin(__builtin_amdgcn_global_load_async_to_lds_b128)
    __builtin_amdgcn_global_load_async_to_lds_b128(
        (as1_v4i)(unsigned long long)g,
        (as3_v4i)(u32)(unsigned long long)l, 0, 0);
#else
    asm volatile("global_load_async_to_lds_b128 %0, %1, off"
                 :: "v"((u32)(unsigned long long)l), "v"(g) : "memory");
#endif
}
#if __has_builtin(__builtin_amdgcn_s_wait_asynccnt)
#define WAIT_ASYNC(n) __builtin_amdgcn_s_wait_asynccnt(n)
#else
#define WAIT_ASYNC(n) asm volatile("s_wait_asynccnt %0" :: "i"(n) : "memory")
#endif

__device__ __forceinline__ ushort_t f2bf_rne(float f) {
    u32 x = __float_as_uint(f);
    u32 r = x + 0x7FFFu + ((x >> 16) & 1u);   // round-to-nearest-even
    return (ushort_t)(r >> 16);
}
__device__ __forceinline__ float bf2f(ushort_t h) {
    return __uint_as_float(((u32)h) << 16);
}
__device__ __forceinline__ u64 shfl_xor_u64(u64 v, int m) {
    u32 lo = (u32)v, hi = (u32)(v >> 32);
    lo = __shfl_xor(lo, m, 32);
    hi = __shfl_xor(hi, m, 32);
    return (((u64)hi) << 32) | (u64)lo;
}

union FragU { v16bf bf; v4u u[2]; };

// ---------------- Kernel 1: row norms, key init, optional bf16 hi/lo planes ----------------
__global__ void __launch_bounds__(256)
rownorm_init_kernel(const float* __restrict__ x, int d,
                    float* __restrict__ sq, u64* __restrict__ pk,
                    u64* __restrict__ nk, float* __restrict__ out,
                    ushort_t* __restrict__ xhi, ushort_t* __restrict__ xlo,
                    int write_bf16) {
    int row = blockIdx.x;
    const float* xr = x + (size_t)row * d;
    float s = 0.f;
    if (write_bf16) {
        for (int i = threadIdx.x; i < d; i += 256) {
            float v = xr[i];
            s += v * v;
            ushort_t h = f2bf_rne(v);
            xhi[(size_t)row * d + i] = h;
            xlo[(size_t)row * d + i] = f2bf_rne(v - bf2f(h));
        }
    } else {
        for (int i = threadIdx.x; i < d; i += 256) { float v = xr[i]; s += v * v; }
    }
    __shared__ float red[256];
    red[threadIdx.x] = s;
    __syncthreads();
    for (int off = 128; off > 0; off >>= 1) {
        if (threadIdx.x < off) red[threadIdx.x] += red[threadIdx.x + off];
        __syncthreads();
    }
    if (threadIdx.x == 0) {
        sq[row] = red[0];
        pk[row] = 0ull;
        nk[row] = ~0ull;
        if (row == 0) out[0] = 0.f;
    }
}

// ---- shared epilogue: distances + packed-key argmax/argmin + per-row atomics ----
__device__ __forceinline__ void mine_epilogue(
        const v8f C[4][2], const float* rsq, const float* csq,
        const int* rid, const int* cid,
        int row0, int col0, int waveM, int waveN, int l15, bool hiHalf,
        u64* __restrict__ pk, u64* __restrict__ nk) {
    #pragma unroll
    for (int mt = 0; mt < 4; ++mt) {
        #pragma unroll
        for (int e = 0; e < 8; ++e) {
            int   mloc = waveM * 64 + mt * 16 + e + (hiHalf ? 8 : 0);
            int   row  = row0 + mloc;
            float rs   = rsq[mloc];
            int   ri   = rid[mloc];
            u64 pkey = 0ull, nkey = ~0ull;
            #pragma unroll
            for (int nt = 0; nt < 2; ++nt) {
                int   cloc = waveN * 32 + nt * 16 + l15;
                u32   col  = (u32)(col0 + cloc);
                float sdot = C[mt][nt][e];
                float d2   = rs + csq[cloc] - 2.0f * sdot;
                float dist = sqrtf(fmaxf(d2, 0.0f));
                bool  same = (ri == cid[cloc]);
                float pv   = same ? dist : 0.0f;
                float nv   = same ? __uint_as_float(0x7F800000u) : dist;
                u64 pkc = (((u64)__float_as_uint(pv)) << 32) | (u64)(0xFFFFFFFFu - col);
                u64 nkc = (((u64)__float_as_uint(nv)) << 32) | (u64)col;
                pkey = pkc > pkey ? pkc : pkey;
                nkey = nkc < nkey ? nkc : nkey;
            }
            #pragma unroll
            for (int m = 1; m < 16; m <<= 1) {   // butterfly within 16-lane half
                u64 p2 = shfl_xor_u64(pkey, m);
                u64 n2 = shfl_xor_u64(nkey, m);
                pkey = p2 > pkey ? p2 : pkey;
                nkey = n2 < nkey ? n2 : nkey;
            }
            if (l15 == 0) {
                atomicMax(&pk[row], pkey);
                atomicMin(&nk[row], nkey);
            }
        }
    }
}

// ---------------- Kernel 2a: async-staged WMMA GEMM + mining (preferred) ----------------
__global__ void __launch_bounds__(256)
gemm_mine_async_kernel(const ushort_t* __restrict__ xhi, const ushort_t* __restrict__ xlo,
                       const int* __restrict__ ident, const float* __restrict__ sq,
                       int d, u64* __restrict__ pk, u64* __restrict__ nk) {
    __shared__ ushort_t SH[2][4][128 * LDT];   // [buf][Ahi,Alo,Bhi,Blo]
    __shared__ float rsq[128], csq[128];
    __shared__ int   rid[128], cid[128];

    const int tid  = threadIdx.x;
    const int row0 = blockIdx.y * 128;
    const int col0 = blockIdx.x * 128;

    if (tid < 128) { rsq[tid] = sq[row0 + tid]; rid[tid] = ident[row0 + tid]; }
    else { int t = tid - 128; csq[t] = sq[col0 + t]; cid[t] = ident[col0 + t]; }

    const int  wave   = tid >> 5;
    const int  lane   = tid & 31;
    const int  waveM  = wave >> 2;
    const int  waveN  = wave & 3;
    const int  l15    = lane & 15;
    const bool hiHalf = lane >= 16;

    // async staging assignment: wave -> (tile, 64-row half); lane -> (row-in-8, 16B chunk)
    const int  stile = wave >> 1;              // 0:Ahi 1:Alo 2:Bhi 3:Blo
    const int  shalf = wave & 1;
    const ushort_t* tsrc = ((stile == 0 || stile == 2) ? xhi : xlo)
                           + (size_t)((stile < 2) ? row0 : col0) * d;
    const int  lrow = shalf * 64 + (lane >> 2);  // 0..127
    const int  lcol = (lane & 3) * 8;            // ushort offset within 32-wide row

    auto issue = [&](int bufi, int k0) {
        const ushort_t* s  = tsrc + (size_t)lrow * d + k0 + lcol;
        ushort_t*       dt = &SH[bufi][stile][lrow * LDT + lcol];
        #pragma unroll
        for (int i = 0; i < 8; ++i) {            // 8 instr x 512B = one 64x32 bf16 half-tile / wave
            async_ld_b128(s, dt);
            s  += (size_t)8 * d;
            dt += 8 * LDT;
        }
    };

    v8f C[4][2] = {};
    const int nchunks = d / KC;

    issue(0, 0);                                  // prologue: chunk 0 -> buf 0
    for (int kc = 0; kc < nchunks; ++kc) {
        const int cur = kc & 1;
        if (kc + 1 < nchunks) {
            issue(cur ^ 1, (kc + 1) * KC);        // prefetch next chunk into other buffer
            WAIT_ASYNC(8);                        // oldest 8 (current chunk) complete
        } else {
            WAIT_ASYNC(0);
        }
        __syncthreads();

        const ushort_t* Ahi = SH[cur][0];
        const ushort_t* Alo = SH[cur][1];
        const ushort_t* Bhi = SH[cur][2];
        const ushort_t* Blo = SH[cur][3];

        FragU ah[4], al[4], bh[2], bl[2];
        const int akb = hiHalf ? 8 : 0;           // A lanes16-31: K=8..15 / 24..31
        #pragma unroll
        for (int mt = 0; mt < 4; ++mt) {
            int r = waveM * 64 + mt * 16 + l15;
            ah[mt].u[0] = *(const v4u*)&Ahi[r * LDT + akb];
            ah[mt].u[1] = *(const v4u*)&Ahi[r * LDT + akb + 16];
            al[mt].u[0] = *(const v4u*)&Alo[r * LDT + akb];
            al[mt].u[1] = *(const v4u*)&Alo[r * LDT + akb + 16];
        }
        const int bkb = hiHalf ? 8 : 0;           // B lanes16-31: K=8..15 / 24..31
        #pragma unroll
        for (int nt = 0; nt < 2; ++nt) {
            int c = waveN * 32 + nt * 16 + l15;
            bh[nt].u[0] = *(const v4u*)&Bhi[c * LDT + bkb];
            bh[nt].u[1] = *(const v4u*)&Bhi[c * LDT + bkb + 16];
            bl[nt].u[0] = *(const v4u*)&Blo[c * LDT + bkb];
            bl[nt].u[1] = *(const v4u*)&Blo[c * LDT + bkb + 16];
        }

        #pragma unroll
        for (int mt = 0; mt < 4; ++mt)
            #pragma unroll
            for (int nt = 0; nt < 2; ++nt) {
                C[mt][nt] = __builtin_amdgcn_wmma_f32_16x16x32_bf16(
                    false, ah[mt].bf, false, bh[nt].bf, (short)0, C[mt][nt], false, false);
                C[mt][nt] = __builtin_amdgcn_wmma_f32_16x16x32_bf16(
                    false, ah[mt].bf, false, bl[nt].bf, (short)0, C[mt][nt], false, false);
                C[mt][nt] = __builtin_amdgcn_wmma_f32_16x16x32_bf16(
                    false, al[mt].bf, false, bh[nt].bf, (short)0, C[mt][nt], false, false);
            }
        __syncthreads();                          // buffer consumed; safe to overwrite next iter
    }

    mine_epilogue(C, rsq, csq, rid, cid, row0, col0, waveM, waveN, l15, hiHalf, pk, nk);
}

// ---------------- Kernel 2b: conversion-in-kernel fallback (small workspace) ----------------
__global__ void __launch_bounds__(256)
gemm_mine_conv_kernel(const float* __restrict__ x, const int* __restrict__ ident,
                      const float* __restrict__ sq, int d,
                      u64* __restrict__ pk, u64* __restrict__ nk) {
    __shared__ ushort_t Ahi[128 * LDT], Alo[128 * LDT];
    __shared__ ushort_t Bhi[128 * LDT], Blo[128 * LDT];
    __shared__ float rsq[128], csq[128];
    __shared__ int   rid[128], cid[128];

    const int tid  = threadIdx.x;
    const int row0 = blockIdx.y * 128;
    const int col0 = blockIdx.x * 128;

    if (tid < 128) { rsq[tid] = sq[row0 + tid]; rid[tid] = ident[row0 + tid]; }
    else { int t = tid - 128; csq[t] = sq[col0 + t]; cid[t] = ident[col0 + t]; }

    const int  wave   = tid >> 5;
    const int  lane   = tid & 31;
    const int  waveM  = wave >> 2;
    const int  waveN  = wave & 3;
    const int  l15    = lane & 15;
    const bool hiHalf = lane >= 16;

    v8f C[4][2] = {};
    const int nchunks = d / KC;
    for (int kc = 0; kc < nchunks; ++kc) {
        const int k0 = kc * KC;
        __syncthreads();
        for (int i = 0; i < 4; ++i) {
            int idx = tid + i * 256;
            int r   = idx >> 3;
            int cv  = idx & 7;
            const float4 fa = *(const float4*)(x + (size_t)(row0 + r) * d + k0 + cv * 4);
            const float4 fb = *(const float4*)(x + (size_t)(col0 + r) * d + k0 + cv * 4);
            int base = r * LDT + cv * 4;
            float av[4] = { fa.x, fa.y, fa.z, fa.w };
            float bv[4] = { fb.x, fb.y, fb.z, fb.w };
            #pragma unroll
            for (int j = 0; j < 4; ++j) {
                ushort_t h = f2bf_rne(av[j]);
                Ahi[base + j] = h;
                Alo[base + j] = f2bf_rne(av[j] - bf2f(h));
                h = f2bf_rne(bv[j]);
                Bhi[base + j] = h;
                Blo[base + j] = f2bf_rne(bv[j] - bf2f(h));
            }
        }
        __syncthreads();

        FragU ah[4], al[4], bh[2], bl[2];
        const int akb = hiHalf ? 8 : 0;
        #pragma unroll
        for (int mt = 0; mt < 4; ++mt) {
            int r = waveM * 64 + mt * 16 + l15;
            ah[mt].u[0] = *(const v4u*)&Ahi[r * LDT + akb];
            ah[mt].u[1] = *(const v4u*)&Ahi[r * LDT + akb + 16];
            al[mt].u[0] = *(const v4u*)&Alo[r * LDT + akb];
            al[mt].u[1] = *(const v4u*)&Alo[r * LDT + akb + 16];
        }
        const int bkb = hiHalf ? 8 : 0;
        #pragma unroll
        for (int nt = 0; nt < 2; ++nt) {
            int c = waveN * 32 + nt * 16 + l15;
            bh[nt].u[0] = *(const v4u*)&Bhi[c * LDT + bkb];
            bh[nt].u[1] = *(const v4u*)&Bhi[c * LDT + bkb + 16];
            bl[nt].u[0] = *(const v4u*)&Blo[c * LDT + bkb];
            bl[nt].u[1] = *(const v4u*)&Blo[c * LDT + bkb + 16];
        }

        #pragma unroll
        for (int mt = 0; mt < 4; ++mt)
            #pragma unroll
            for (int nt = 0; nt < 2; ++nt) {
                C[mt][nt] = __builtin_amdgcn_wmma_f32_16x16x32_bf16(
                    false, ah[mt].bf, false, bh[nt].bf, (short)0, C[mt][nt], false, false);
                C[mt][nt] = __builtin_amdgcn_wmma_f32_16x16x32_bf16(
                    false, ah[mt].bf, false, bl[nt].bf, (short)0, C[mt][nt], false, false);
                C[mt][nt] = __builtin_amdgcn_wmma_f32_16x16x32_bf16(
                    false, al[mt].bf, false, bh[nt].bf, (short)0, C[mt][nt], false, false);
            }
    }

    mine_epilogue(C, rsq, csq, rid, cid, row0, col0, waveM, waveN, l15, hiHalf, pk, nk);
}

// ---------------- Kernel 3: exact pairwise_distance recompute + loss ----------------
__global__ void __launch_bounds__(256)
finish_kernel(const float* __restrict__ x, int d,
              const u64* __restrict__ pk, const u64* __restrict__ nk,
              float* __restrict__ out, float invn) {
    int row = blockIdx.x;
    u32 pidx = 0xFFFFFFFFu - (u32)(pk[row] & 0xFFFFFFFFull);
    u32 nidx = (u32)(nk[row] & 0xFFFFFFFFull);
    const float* xr = x + (size_t)row  * d;
    const float* xp = x + (size_t)pidx * d;
    const float* xn = x + (size_t)nidx * d;
    float sp = 0.f, sn = 0.f;
    for (int i = threadIdx.x; i < d; i += 256) {
        float a  = xr[i];
        float dp = a - xp[i] + 1e-6f;
        float dn = a - xn[i] + 1e-6f;
        sp += dp * dp;
        sn += dn * dn;
    }
    __shared__ float rp[256], rn[256];
    rp[threadIdx.x] = sp; rn[threadIdx.x] = sn;
    __syncthreads();
    for (int off = 128; off > 0; off >>= 1) {
        if (threadIdx.x < off) {
            rp[threadIdx.x] += rp[threadIdx.x + off];
            rn[threadIdx.x] += rn[threadIdx.x + off];
        }
        __syncthreads();
    }
    if (threadIdx.x == 0) {
        float l = fmaxf(0.f, 0.2f + sqrtf(rp[0]) - sqrtf(rn[0]));
        atomicAdd(out, l * invn);
    }
}

extern "C" void kernel_launch(void* const* d_in, const int* in_sizes, int n_in,
                              void* d_out, int out_size, void* d_ws, size_t ws_size,
                              hipStream_t stream) {
    const float* feat  = (const float*)d_in[0];
    const int*   ident = (const int*)d_in[1];
    float*       out   = (float*)d_out;

    const int n = in_sizes[1];          // 4096
    const int d = in_sizes[0] / n;      // 2048

    float* sq = (float*)d_ws;
    u64*   pk = (u64*)((char*)d_ws + (size_t)n * sizeof(float));
    u64*   nk = pk + n;
    size_t base = (size_t)n * sizeof(float) + 2ull * n * sizeof(u64);
    ushort_t* xhi = (ushort_t*)((char*)d_ws + base);
    ushort_t* xlo = xhi + (size_t)n * d;
    size_t need = base + 2ull * (size_t)n * d * sizeof(ushort_t);

    const bool use_async = (ws_size >= need);
    dim3 grid(n / 128, n / 128);

    rownorm_init_kernel<<<n, 256, 0, stream>>>(feat, d, sq, pk, nk, out,
                                               xhi, xlo, use_async ? 1 : 0);
    if (use_async) {
        gemm_mine_async_kernel<<<grid, 256, 0, stream>>>(xhi, xlo, ident, sq, d, pk, nk);
    } else {
        gemm_mine_conv_kernel<<<grid, 256, 0, stream>>>(feat, ident, sq, d, pk, nk);
    }
    finish_kernel<<<n, 256, 0, stream>>>(feat, d, pk, nk, out, 1.0f / (float)n);
}
